// BiLSTMModel_68899865362873
// MI455X (gfx1250) — compile-verified, matrix operand order checked by reading, not run on hardware
//
#include <hip/hip_runtime.h>
#include <hip/hip_bf16.h>
#include <math.h>

// ---------------------------------------------------------------------------
// BiLSTM (2 layers, B=64, T=1024, E=256, H=256) for gfx1250 (MI455X).
// Recurrence: v_wmma_f32_16x16x32_bf16, 16 wave32 / block, one hidden group
// (16 cols x 4 gates) per wave, c-state in VGPRs, explicit K ping-pong
// pipeline.  Activation panel DMA'd into LDS by the TDM with row padding so
// it lands directly in the [16][Ktot] A-panel layout; double buffered and
// overlapped with the WMMA K-loop (TENSORcnt).
// ---------------------------------------------------------------------------

typedef __bf16 bf16_t;
typedef __bf16 v16bf __attribute__((ext_vector_type(16)));
typedef float  v8f   __attribute__((ext_vector_type(8)));
typedef unsigned int v4u __attribute__((ext_vector_type(4)));
typedef int v4i __attribute__((ext_vector_type(4)));
typedef int v8i __attribute__((ext_vector_type(8)));

#define BB 64
#define TT 1024
#define EE 256
#define HH 256
#define NG 1024    // 4*H gate width
#define NTHR 512   // 16 wave32 per scan block
#define KMAX 768   // max Ktot (layer 2)

#ifndef __has_builtin
#define __has_builtin(x) 0
#endif
#if __has_builtin(__builtin_amdgcn_tensor_load_to_lds) && \
    __has_builtin(__builtin_amdgcn_s_wait_tensorcnt)
#define HAVE_TDM 1
#warning "PROBE: TDM path ENABLED"
#else
#define HAVE_TDM 0
#warning "PROBE: TDM path DISABLED (fallback cooperative copy)"
#endif

// ---------------------------------------------------------------------------
// Embedding gather + fp32 -> bf16 conversion.
// ---------------------------------------------------------------------------
__global__ void embed_gather_kernel(const int* __restrict__ tokens,
                                    const float* __restrict__ embed,
                                    bf16_t* __restrict__ X1) {
  const int bt = blockIdx.x;           // 0 .. B*T-1
  const int e  = threadIdx.x;          // 0 .. E-1
  const int tok = tokens[bt];
  X1[(size_t)bt * EE + e] = (bf16_t)embed[(size_t)tok * EE + e];
}

// ---------------------------------------------------------------------------
// Pack [Wi; Wh] (row-major [K x 4H], fp32) into transposed bf16 [4H x K].
// ---------------------------------------------------------------------------
__global__ void pack_weights_kernel(const float* __restrict__ Wi,
                                    const float* __restrict__ Wh,
                                    bf16_t* __restrict__ Wt,
                                    int Din) {
  const int Ktot = Din + HH;
  const size_t idx = (size_t)blockIdx.x * blockDim.x + threadIdx.x;
  const size_t total = (size_t)NG * (size_t)Ktot;
  if (idx >= total) return;
  const int n = (int)(idx / Ktot);
  const int k = (int)(idx % Ktot);
  const float v = (k < Din) ? Wi[(size_t)k * NG + n]
                            : Wh[(size_t)(k - Din) * NG + n];
  Wt[idx] = (bf16_t)v;
}

__device__ __forceinline__ float sigf(float x) {
  return 1.0f / (1.0f + __expf(-x));
}

#if HAVE_TDM
// TDM 2D tile load: 16 rows x din_elems bf16 (row stride row_stride_elems in
// memory) into LDS at lds_off with HH*2 bytes of padding appended to each row
// so the tile lands in a [16][Din+HH] panel.  D# per ISA §8.3/§8.4.
__device__ __forceinline__ void tdm_issue_load(const void* gaddr,
                                               unsigned lds_off,
                                               int din_elems,
                                               unsigned long long row_stride_elems) {
  const unsigned long long ga = (unsigned long long)(uintptr_t)gaddr;
  v4u g0;
  g0.x = 1u;                                         // count=1, user mode
  g0.y = lds_off;                                    // lds_addr
  g0.z = (unsigned)(ga & 0xFFFFFFFFu);               // global_addr[31:0]
  g0.w = (unsigned)((ga >> 32) & 0x1FFFFFFu)         // global_addr[56:32]
         | (2u << 30);                               // type=2 ("image")
  // pad_interval: every Din*2 bytes; codes: 6 -> 128 DW (512B), 7 -> 256 DW.
  const unsigned padint = (din_elems == 256) ? 6u : 7u;
  const unsigned padamt = 127u;                      // 128 DW = 512 B = HH*2
  v8i g1;
  const unsigned td0 = (unsigned)din_elems;          // tensor_dim0
  const unsigned td1 = (unsigned)(BB * TT);          // tensor_dim1 (rows)
  g1[0] = (int)((1u << 16) |                         // data_size = 2 bytes
                (1u << 20) |                         // pad_enable
                (padint << 22) | (padamt << 25));
  g1[1] = (int)((td0 & 0xFFFFu) << 16);              // tensor_dim0[15:0]
  g1[2] = (int)((td0 >> 16) | ((td1 & 0xFFFFu) << 16));
  g1[3] = (int)((td1 >> 16) | ((unsigned)din_elems << 16));  // tile_dim0
  g1[4] = 16;                                        // tile_dim1=16, tile_dim2=0
  g1[5] = (int)(row_stride_elems & 0xFFFFFFFFull);   // dim0_stride[31:0]
  g1[6] = (int)((row_stride_elems >> 32) & 0xFFFFull);
  g1[7] = 0;
  const v4i z4 = {0, 0, 0, 0};
#if __has_include(<hip/amd_detail/amd_gfx1250_TDM.h>)
  const v8i z8 = {0, 0, 0, 0, 0, 0, 0, 0};
  __builtin_amdgcn_tensor_load_to_lds(g0, g1, z4, z4, z8, 0);
#else
  __builtin_amdgcn_tensor_load_to_lds(g0, g1, z4, z4, 0);
#endif
}
#endif

// ---------------------------------------------------------------------------
// Fused LSTM time scan.  grid = (B/16 batch tiles, 2 directions), 512 thr.
// Wave w (0..15) owns hidden group w: gate tiles at N = q*256 + w*16.
// ---------------------------------------------------------------------------
__global__ __launch_bounds__(NTHR) void lstm_scan_kernel(
    const bf16_t* __restrict__ X, int Din,
    const bf16_t* __restrict__ Wt_f, const bf16_t* __restrict__ Wt_b,
    const float* __restrict__ bias_f, const float* __restrict__ bias_b,
    bf16_t* __restrict__ Y, int y_stride, int steps_rev) {

  const int Ktot = Din + HH;                 // 512 (layer1) or 768 (layer2)
  const int b0   = blockIdx.x * 16;          // batch-tile origin
  const int rev  = blockIdx.y;               // 0 = forward, 1 = backward
  const bf16_t* __restrict__ Wt   = rev ? Wt_b  : Wt_f;
  const float*  __restrict__ bias = rev ? bias_b : bias_f;
  const int coloff = rev ? HH : 0;
  const int steps  = rev ? steps_rev : TT;

  // Double-buffered unified A panel: [buf][16 rows][Ktot] (x cols | h cols).
  __shared__ bf16_t As[2][16 * KMAX];        // 48 KB

  const int tid   = threadIdx.x;
  const int wave  = tid >> 5;                // 0..15 == hidden group
  const int lane  = tid & 31;
  const int ln16  = lane & 15;
  const int khalf = (lane >> 4) * 16;        // A/B fragment K half select
  const int mrow  = (lane >> 4) * 8;         // C/D fragment row offset
  const int hcol  = wave * 16 + ln16;        // this lane's hidden column
  const int arow  = ln16 * Ktot + khalf;     // this lane's A-row base (elems)

  // h_{-1} = 0 in both buffers.
  for (int i = tid; i < 16 * HH; i += NTHR) {
    const int r = i / HH, c = i % HH;
    As[0][r * Ktot + Din + c] = (bf16_t)0.0f;
    As[1][r * Ktot + Din + c] = (bf16_t)0.0f;
  }

  // Per-gate byte offsets into Wt (SGPR base + 32-bit lane offset form).
  unsigned boff[4];
#pragma unroll
  for (int q = 0; q < 4; ++q)
    boff[q] = (unsigned)(((q * HH + hcol) * Ktot + khalf) * 2);

  // Bias is time-invariant.
  float bv[4];
#pragma unroll
  for (int q = 0; q < 4; ++q) bv[q] = bias[q * HH + hcol];

  // c state in VGPRs for the whole scan.
  float cst[8];
#pragma unroll
  for (int e = 0; e < 8; ++e) cst[e] = 0.0f;

  auto x_src = [&](int xpos) {
    return X + ((size_t)b0 * TT + (size_t)xpos) * Din;
  };
#if HAVE_TDM
  auto prefetch_x = [&](int buf, int xpos) {
    if (wave == 0)
      tdm_issue_load(x_src(xpos), (unsigned)(uintptr_t)&As[buf][0], Din,
                     (unsigned long long)TT * (unsigned long long)Din);
  };
  auto wait_x = [&]() {
    if (wave == 0) __builtin_amdgcn_s_wait_tensorcnt((short)0);
  };
#else
  auto prefetch_x = [&](int buf, int xpos) {
    const bf16_t* src = x_src(xpos);
    const int chunks = (16 * Din) / 8;       // 16-byte chunks
    for (int c = tid; c < chunks; c += NTHR) {
      const int r   = (c * 8) / Din;
      const int col = (c * 8) % Din;
      *(uint4*)(&As[buf][r * Ktot + col]) =
          *(const uint4*)(src + (size_t)r * TT * Din + col);
    }
  };
  auto wait_x = [&]() {};
#endif

  prefetch_x(0, rev ? (TT - 1) : 0);

  const int nk = Ktot / 32;                  // 16 or 24 (even)

  for (int t = 0; t < steps; ++t) {
    const int xpos = rev ? (TT - 1 - t) : t;
    const int cur  = t & 1;

    wait_x();            // TDM for As[cur] x-region complete
    __syncthreads();     // As[cur] (x + h_{t-1}) visible to all waves

    if (t + 1 < steps)   // fill As[cur^1] x-region during our K-loop
      prefetch_x(cur ^ 1, rev ? (TT - 2 - t) : (t + 1));

    v8f acc[4];
#pragma unroll
    for (int q = 0; q < 4; ++q) {
      v8f a;
#pragma unroll
      for (int e = 0; e < 8; ++e) a[e] = bv[q];
      acc[q] = a;
    }

    const bf16_t* Apanel = &As[cur][0];
    auto loadA = [&](int ki) {
      return *(const v16bf*)(Apanel + arow + ki * 32);
    };
    auto loadB = [&](int ki, int q) {
      return *(const v16bf*)((const char*)Wt + boff[q] + (unsigned)ki * 64u);
    };

    // Explicit ping-pong K pipeline: fetch k+1 fragments while WMMAing k.
    v16bf aA = loadA(0);
    v16bf bA[4];
#pragma unroll
    for (int q = 0; q < 4; ++q) bA[q] = loadB(0, q);

    for (int ki = 0; ki < nk; ki += 2) {
      v16bf aB = loadA(ki + 1);
      v16bf bB[4];
#pragma unroll
      for (int q = 0; q < 4; ++q) bB[q] = loadB(ki + 1, q);
#pragma unroll
      for (int q = 0; q < 4; ++q)
        acc[q] = __builtin_amdgcn_wmma_f32_16x16x32_bf16(
            false, aA, false, bA[q], (short)0, acc[q], false, false);
      if (ki + 2 < nk) {
        aA = loadA(ki + 2);
#pragma unroll
        for (int q = 0; q < 4; ++q) bA[q] = loadB(ki + 2, q);
      }
#pragma unroll
      for (int q = 0; q < 4; ++q)
        acc[q] = __builtin_amdgcn_wmma_f32_16x16x32_bf16(
            false, aB, false, bB[q], (short)0, acc[q], false, false);
    }

    __syncthreads();     // all LDS reads of this step are done

    // Gates: i,f,g,o share the same lane/VGPR slot across the 4 gate tiles.
#pragma unroll
    for (int e = 0; e < 8; ++e) {
      const float iv = acc[0][e];
      const float fv = acc[1][e];
      const float gv = acc[2][e];
      const float ov = acc[3][e];
      const float cnew = sigf(fv) * cst[e] + sigf(iv) * tanhf(gv);
      const float hnew = sigf(ov) * tanhf(cnew);
      cst[e] = cnew;
      const int m = e + mrow;
      const bf16_t hb = (bf16_t)hnew;
      As[0][m * Ktot + Din + hcol] = hb;   // h feeds both panel buffers
      As[1][m * Ktot + Din + hcol] = hb;
      Y[((size_t)(b0 + m) * TT + xpos) * y_stride + coloff + hcol] = hb;
    }
    // Next iteration's first __syncthreads publishes the h writes.
  }
}

// ---------------------------------------------------------------------------
// Head: out[b] = sigmoid(X3[b, T-1, :] . Wd + bd)
// ---------------------------------------------------------------------------
__global__ void dense_head_kernel(const bf16_t* __restrict__ X3,
                                  const float* __restrict__ Wd,
                                  const float* __restrict__ bd,
                                  float* __restrict__ out) {
  const int b = blockIdx.x;
  __shared__ float red[256];
  float s = 0.0f;
  for (int j = threadIdx.x; j < 2 * HH; j += 256)
    s += (float)X3[((size_t)b * TT + (TT - 1)) * (2 * HH) + j] * Wd[j];
  red[threadIdx.x] = s;
  __syncthreads();
  for (int off = 128; off > 0; off >>= 1) {
    if (threadIdx.x < off) red[threadIdx.x] += red[threadIdx.x + off];
    __syncthreads();
  }
  if (threadIdx.x == 0) out[b] = 1.0f / (1.0f + __expf(-(red[0] + bd[0])));
}

// ---------------------------------------------------------------------------
// Launcher
// ---------------------------------------------------------------------------
extern "C" void kernel_launch(void* const* d_in, const int* in_sizes, int n_in,
                              void* d_out, int out_size, void* d_ws,
                              size_t ws_size, hipStream_t stream) {
  const int*   tokens = (const int*)d_in[0];
  const float* embed  = (const float*)d_in[1];
  const float* fw1_Wi = (const float*)d_in[2];
  const float* fw1_Wh = (const float*)d_in[3];
  const float* fw1_b  = (const float*)d_in[4];
  const float* bw1_Wi = (const float*)d_in[5];
  const float* bw1_Wh = (const float*)d_in[6];
  const float* bw1_b  = (const float*)d_in[7];
  const float* fw2_Wi = (const float*)d_in[8];
  const float* fw2_Wh = (const float*)d_in[9];
  const float* fw2_b  = (const float*)d_in[10];
  const float* bw2_Wi = (const float*)d_in[11];
  const float* bw2_Wh = (const float*)d_in[12];
  const float* bw2_b  = (const float*)d_in[13];
  const float* Wd     = (const float*)d_in[14];
  const float* bd     = (const float*)d_in[15];
  float* out = (float*)d_out;

  // Workspace layout (bf16 buffers), ~165 MB total.
  char* ws = (char*)d_ws;
  const size_t szX1 = (size_t)BB * TT * EE * 2;        // 32 MB
  const size_t szX2 = (size_t)BB * TT * 2 * HH * 2;    // 64 MB
  const size_t szX3 = szX2;                            // 64 MB
  const size_t szW1 = (size_t)NG * (EE + HH) * 2;      // 1 MB
  const size_t szW2 = (size_t)NG * (2 * HH + HH) * 2;  // 1.5 MB

  bf16_t* X1  = (bf16_t*)(ws);
  bf16_t* X2  = (bf16_t*)(ws + szX1);
  bf16_t* X3  = (bf16_t*)(ws + szX1 + szX2);
  bf16_t* W1f = (bf16_t*)(ws + szX1 + szX2 + szX3);
  bf16_t* W1b = (bf16_t*)(ws + szX1 + szX2 + szX3 + szW1);
  bf16_t* W2f = (bf16_t*)(ws + szX1 + szX2 + szX3 + 2 * szW1);
  bf16_t* W2b = (bf16_t*)(ws + szX1 + szX2 + szX3 + 2 * szW1 + szW2);

  // 1) Embedding gather -> bf16
  embed_gather_kernel<<<BB * TT, EE, 0, stream>>>(tokens, embed, X1);

  // 2) Pack weights: [Wi;Wh] -> transposed bf16 [4H x K]
  {
    const int K1 = EE + HH, K2 = 2 * HH + HH;
    const int n1 = (NG * K1 + 255) / 256;
    const int n2 = (NG * K2 + 255) / 256;
    pack_weights_kernel<<<n1, 256, 0, stream>>>(fw1_Wi, fw1_Wh, W1f, EE);
    pack_weights_kernel<<<n1, 256, 0, stream>>>(bw1_Wi, bw1_Wh, W1b, EE);
    pack_weights_kernel<<<n2, 256, 0, stream>>>(fw2_Wi, fw2_Wh, W2f, 2 * HH);
    pack_weights_kernel<<<n2, 256, 0, stream>>>(bw2_Wi, bw2_Wh, W2b, 2 * HH);
  }

  // 3) BiLSTM layer 1: X1 [B,T,256] -> X2 [B,T,512] (full scans both dirs)
  lstm_scan_kernel<<<dim3(BB / 16, 2), NTHR, 0, stream>>>(
      X1, EE, W1f, W1b, fw1_b, bw1_b, X2, 2 * HH, TT);

  // 4) BiLSTM layer 2: X2 -> X3.  Head reads only t=T-1; the backward
  //    LSTM's value there is its FIRST scan step -> 1 step only.
  lstm_scan_kernel<<<dim3(BB / 16, 2), NTHR, 0, stream>>>(
      X2, 2 * HH, W2f, W2b, fw2_b, bw2_b, X3, 2 * HH, 1);

  // 5) Dense head + sigmoid on the last timestep
  dense_head_kernel<<<BB, 256, 0, stream>>>(X3, Wd, bd, out);
}
